// MLPSCM_2748779069619
// MI455X (gfx1250) — matrix operand order repaired and checked by reading, not compiled, not run on hardware
//
#include <hip/hip_runtime.h>
#include <hip/hip_bf16.h>
#include <math.h>

#define S_LEN 131072
#define NN 256
#define NC 16
#define NF 100
#define NSTEP (NN - NC) // 240

typedef __bf16 bf16_t;
typedef bf16_t v16bf __attribute__((ext_vector_type(16)));
typedef float v8f __attribute__((ext_vector_type(8)));

union FragAB {
    v16bf v;
    uint4 u[2];
};

// d_ws layout:
//   [0,       131072)  : WB   bf16[256][256], WB[j][k]  = W[k][j]*mask[k][j] (transposed, bf16)
//   [131072,  393216)  : WfT  f32 [256][256], WfT[j][k] = W[k][j]*mask[k][j] (transposed, f32)
//   [393216,  394240)  : feat int[256], feature index of node or -1
//   [394240,  394244)  : NaN flag (int)
#define WS_WFT 131072
#define WS_FEAT 393216
#define WS_FLAG 394240

__global__ void prep_kernel(const float* __restrict__ W, const int* __restrict__ mask,
                            const int* __restrict__ X_idx, char* __restrict__ ws) {
    bf16_t* WB  = (bf16_t*)ws;
    float* WfT  = (float*)(ws + WS_WFT);
    int* feat   = (int*)(ws + WS_FEAT);
    int* flag   = (int*)(ws + WS_FLAG);
    int j = blockIdx.x;
    int k = threadIdx.x;
    if (j < NN) {
        float wv = W[k * NN + j] * (mask[k * NN + j] ? 1.0f : 0.0f);
        WB[j * NN + k]  = (bf16_t)wv;
        WfT[j * NN + k] = wv;
    } else {
        feat[k] = -1;
        __syncthreads();
        if (k < NF) feat[X_idx[k]] = k;
        if (k == 0) *flag = 0;
    }
}

// 2 waves (64 threads) per block; each wave owns 32 consecutive samples
// (two 16-row WMMA M-tiles sharing one B fragment).
__global__ __launch_bounds__(64) void scm_kernel(
    const float* __restrict__ causes, const float* __restrict__ eps,
    const int* __restrict__ act_id, const int* __restrict__ y_idx,
    char* __restrict__ ws, float* __restrict__ Xout, float* __restrict__ yout) {

    const bf16_t* WB = (const bf16_t*)ws;
    const float* WfT = (const float*)(ws + WS_WFT);
    const int* feat  = (const int*)(ws + WS_FEAT);
    int* flag        = (int*)(ws + WS_FLAG);

    // vals rows padded 256 -> 264 bf16 to spread LDS banks; zl rows 16 -> 17 f32.
    __shared__ alignas(16) bf16_t vals[2][32][264];
    __shared__ float zl[2][32][17];

    const int t = threadIdx.x & 31;       // lane
    const int w = threadIdx.x >> 5;       // wave in block
    const int sbase = blockIdx.x * 64 + w * 32;
    const int sample = sbase + t;
    const int lohalf = (t >> 4) & 1;      // lane half (ISA wave32 fragment layouts)
    const int nloc = t & 15;

    // ---- init: causes into cols 0..15 (bf16), zero cols 16..263 ----
    {
        const float4* cz = (const float4*)(causes + (size_t)sample * NC);
        float4 c0 = cz[0], c1 = cz[1], c2 = cz[2], c3 = cz[3];
        bf16_t* row = vals[w][t];
        row[0]=(bf16_t)c0.x;  row[1]=(bf16_t)c0.y;  row[2]=(bf16_t)c0.z;  row[3]=(bf16_t)c0.w;
        row[4]=(bf16_t)c1.x;  row[5]=(bf16_t)c1.y;  row[6]=(bf16_t)c1.z;  row[7]=(bf16_t)c1.w;
        row[8]=(bf16_t)c2.x;  row[9]=(bf16_t)c2.y;  row[10]=(bf16_t)c2.z; row[11]=(bf16_t)c2.w;
        row[12]=(bf16_t)c3.x; row[13]=(bf16_t)c3.y; row[14]=(bf16_t)c3.z; row[15]=(bf16_t)c3.w;
        uint4 z4 = {0u, 0u, 0u, 0u};
        uint4* p = (uint4*)(row + 16);    // 32B offset, 16B aligned
        #pragma unroll
        for (int i = 0; i < 31; ++i) p[i] = z4;
    }
    asm volatile("s_wait_dscnt 0x0" ::: "memory"); // wave-lockstep: slab ready

    float v[16];
    bool nanbad = false;
    const int yj = y_idx[0];

    for (int b = 1; b <= 15; ++b) {
        const int j0 = b * 16;
        v8f acc0 = {};
        v8f acc1 = {};
        const int nch = (j0 + 31) >> 5;   // K chunks of 32; partial chunk reads zeros

        for (int c = 0; c < nch; ++c) {
            const int kb = c * 32;
            // B fragment: 32x16 bf16. Lane half selects K 0..15 / 16..31, n = lane&15.
            FragAB fb;
            const bf16_t* bptr = WB + (size_t)(j0 + nloc) * NN + kb + lohalf * 16;
            fb.u[0] = ((const uint4*)bptr)[0];
            fb.u[1] = ((const uint4*)bptr)[1];
            // A fragments: 16x32 bf16 per M-tile. u[0]=K ka..ka+7, u[1]=K ka+8..ka+15.
            const int ka = kb + lohalf * 16;
            FragAB fa0, fa1;
            const bf16_t* a0p = &vals[w][nloc][ka];
            fa0.u[0] = *(const uint4*)a0p;
            fa0.u[1] = *(const uint4*)(a0p + 8);
            const bf16_t* a1p = &vals[w][16 + nloc][ka];
            fa1.u[0] = *(const uint4*)a1p;
            fa1.u[1] = *(const uint4*)(a1p + 8);

            acc0 = __builtin_amdgcn_wmma_f32_16x16x32_bf16(false, fa0.v, false, fb.v,
                                                           (short)0, acc0, false, false);
            acc1 = __builtin_amdgcn_wmma_f32_16x16x32_bf16(false, fa1.v, false, fb.v,
                                                           (short)0, acc1, false, false);
        }

        // C/D layout -> zl: VGPR r holds M = r + 8*lanehalf, N = lane&15.
        #pragma unroll
        for (int r = 0; r < 8; ++r) {
            zl[w][r + lohalf * 8][nloc]      = acc0[r];
            zl[w][16 + r + lohalf * 8][nloc] = acc1[r];
        }
        asm volatile("s_wait_dscnt 0x0" ::: "memory");

        // eps tile for this sample (row-contiguous, 64B aligned)
        const float4* ep = (const float4*)(eps + (size_t)sample * NSTEP + (j0 - NC));
        float4 e0 = ep[0], e1 = ep[1], e2 = ep[2], e3 = ep[3];
        float ev[16] = {e0.x, e0.y, e0.z, e0.w, e1.x, e1.y, e1.z, e1.w,
                        e2.x, e2.y, e2.z, e2.w, e3.x, e3.y, e3.z, e3.w};
        if (b < 15)
            __builtin_prefetch(eps + (size_t)sample * NSTEP + j0, 0, 1);

        // in-block triangular resolve: lane t owns sample t of the wave slab
        #pragma unroll
        for (int i = 0; i < 16; ++i) {
            const int j = j0 + i;
            float z = zl[w][t][i] + 0.01f * ev[i];
            const float* wrow = WfT + (size_t)j * NN + j0; // uniform -> scalar loads
            #pragma unroll
            for (int l = 0; l < i; ++l) z += wrow[l] * v[l];
            const int a = act_id[j - NC];
            float val;
            if (a == 0)      val = z;
            else if (a == 1) val = tanhf(z);
            else if (a == 2) val = fmaxf(z, 0.0f);
            else             val = 1.0f / (1.0f + __expf(-z));
            v[i] = val;
            vals[w][t][j] = (bf16_t)val;
            const int f = feat[j];                  // uniform
            if (f >= 0) Xout[(size_t)sample * NF + f] = val; // L2-coalesced over time
            if (j == yj) yout[sample] = val;
            nanbad |= (val != val);
        }
        asm volatile("s_wait_dscnt 0x0" ::: "memory"); // new bf16 vals visible to next GEMM
    }

    if (nanbad) atomicOr(flag, 1);
}

__global__ void nanfix_kernel(const char* __restrict__ ws, float* __restrict__ out,
                              size_t nX, size_t total) {
    const int* flag = (const int*)(ws + WS_FLAG);
    if (*flag == 0) return;
    size_t i = (size_t)blockIdx.x * blockDim.x + threadIdx.x;
    if (i >= total) return;
    out[i] = (i < nX) ? 0.0f : -100.0f;
}

extern "C" void kernel_launch(void* const* d_in, const int* in_sizes, int n_in,
                              void* d_out, int out_size, void* d_ws, size_t ws_size,
                              hipStream_t stream) {
    (void)in_sizes; (void)n_in; (void)out_size; (void)ws_size;
    const float* causes = (const float*)d_in[0];
    const float* W      = (const float*)d_in[1];
    const float* eps    = (const float*)d_in[2];
    const int*   mask   = (const int*)d_in[3];
    const int*   act    = (const int*)d_in[4];
    const int*   Xidx   = (const int*)d_in[5];
    const int*   yidx   = (const int*)d_in[6];
    float* out = (float*)d_out;
    char*  ws  = (char*)d_ws;

    prep_kernel<<<NN + 1, NN, 0, stream>>>(W, mask, Xidx, ws);
    scm_kernel<<<S_LEN / 64, 64, 0, stream>>>(causes, eps, act, yidx, ws,
                                              out, out + (size_t)S_LEN * NF);
    const size_t nX = (size_t)S_LEN * NF;
    const size_t total = nX + S_LEN;
    nanfix_kernel<<<(unsigned)((total + 255) / 256), 256, 0, stream>>>(ws, out, nX, total);
}
